// DCGRUEncoder_86285892976921
// MI455X (gfx1250) — matrix operand order repaired and checked by reading, not compiled, not run on hardware
//
#include <hip/hip_runtime.h>
#include <cstdint>
#include <cstddef>

// ---------------- problem constants (from reference) ----------------
#define DC_T 12
#define DC_B 64
#define DC_N 1024
#define DC_I 2
#define DC_H 128

// ---------------- GEMM tiling ----------------
#define MBLK 128
#define NBLK 128
#define KBLK 32
#define LDSBUF 4096   // u32 elements per double-buffer half (As 2048 + Bs 2048)

typedef unsigned short u16;
typedef unsigned int   u32;
typedef __attribute__((ext_vector_type(16))) __bf16 v16bf;
typedef __attribute__((ext_vector_type(8)))  __bf16 v8bf;
typedef __attribute__((ext_vector_type(8)))  float  v8f;

__device__ __forceinline__ u16 f2bf(float f) {
    union { float f; u32 u; } v; v.f = f;
    u32 r = v.u + 0x7FFFu + ((v.u >> 16) & 1u);   // round-to-nearest-even
    return (u16)(r >> 16);
}
__device__ __forceinline__ float bf2f(u16 b) {
    union { u32 u; float f; } v; v.u = ((u32)b) << 16;
    return v.f;
}

// ---------------------------------------------------------------------------
// Specialized bf16 WMMA GEMM:  out = alpha*(A@B) [+ cbeta*Cadd] [+ bias(col)]
//   NSEG : 1 (plain A) or 3 (A = [A0|A1|A2] along K, each (M,segK) row-major)
//   FULLK: Ktot % 32 == 0 (no K-tail guards)
//   CADD : 0 none, 1 f32 Cadd, 2 bf16 Cadd;  OUTF/OUTB: f32/bf16 result;  BIAS
// Host guarantees: M%128==0, Nc%128==0, segK%8==0, A buffers have tail slack
// (K-tail A reads hit slack garbage that multiplies zero-filled B rows).
// 256 threads = 8 wave32 (4x2); wave owns 32x64 = 2x4 WMMA accumulators.
// Double-buffered LDS (2 x 16KB), software-pipelined: one barrier per k-tile,
// global loads for tile i+1 issued before the WMMAs of tile i.
// ---------------------------------------------------------------------------
template<int NSEG, bool FULLK, int CADD, bool OUTF, bool OUTB, bool BIAS>
__global__ __launch_bounds__(256) void gemm_ker(
    const u16* __restrict__ A0, const u16* __restrict__ A1, const u16* __restrict__ A2,
    int segK, int Ktot,
    const u16* __restrict__ Bm, int ldb,
    float* __restrict__ Cf, u16* __restrict__ Cb, int ldc,
    float alpha, const float* __restrict__ CaddF, const u16* __restrict__ CaddB,
    float cbeta, const float* __restrict__ bias)
{
    __shared__ u32 smem[2 * LDSBUF];  // [buf][As 2048 | Bs 2048], chunk-swizzled

    const int tid  = threadIdx.x;
    const int lane = tid & 31;
    const int wave = tid >> 5;
    const int wr = wave >> 1, wc = wave & 1;
    const int half = lane >> 4, lm = lane & 15;

    const int rowBase = blockIdx.y * MBLK;
    const int colBase = blockIdx.x * NBLK;

    // staging thread mapping
    const int arow = tid >> 2;            // 0..63 (+64)
    const int akb  = (tid & 3) << 3;      // 0,8,16,24
    const int achunk = akb >> 3;
    const int bn8  = (tid & 15) << 3;     // 0..120
    const int bkp  = tid >> 4;            // 0..15
    const int bchunk = bkp >> 2, bo = bkp & 3;

    // ---- hoisted A pointers (include row and akb) ----
    const u16* ap0[2]; const u16* ap1[2]; const u16* ap2[2];
#pragma unroll
    for (int g = 0; g < 2; ++g) {
        const size_t rb = (size_t)(rowBase + arow + 64 * g) * segK + akb;
        ap0[g] = A0 + rb; ap1[g] = A1 + rb; ap2[g] = A2 + rb;
    }
    // ---- hoisted B pointer (advanced per staged tile) ----
    const u16* bp = Bm + (size_t)(2 * bkp) * ldb + colBase + bn8;

    // ---- hoisted LDS indices (u32 units, relative to buffer base) ----
    int asIdx[2];
#pragma unroll
    for (int g = 0; g < 2; ++g) {
        const int row = arow + 64 * g;
        asIdx[g] = row * 16 + ((((row >> 2) + achunk) & 3) << 2);
    }
    int bstIdx[4][2];
#pragma unroll
    for (int j = 0; j < 4; ++j) {
        const int n0 = bn8 + 2 * j, n1 = n0 + 1;
        bstIdx[j][0] = 2048 + n0 * 16 + (((((n0 >> 2) ^ (n0 >> 3)) + bchunk) & 3) << 2) + bo;
        bstIdx[j][1] = 2048 + n1 * 16 + (((((n1 >> 2) ^ (n1 >> 3)) + bchunk) & 3) << 2) + bo;
    }
    int afrIdx[2][2];
#pragma unroll
    for (int mt = 0; mt < 2; ++mt) {
        const int m = wr * 32 + mt * 16 + lm;
        const int key = m >> 2;
        afrIdx[mt][0] = m * 16 + (((half + key) & 3) << 2);
        afrIdx[mt][1] = m * 16 + (((2 + half + key) & 3) << 2);
    }
    int bfrIdx[4][2];
#pragma unroll
    for (int nt = 0; nt < 4; ++nt) {
        const int n = wc * 64 + nt * 16 + lm;
        const int key = (n >> 2) ^ (n >> 3);
        bfrIdx[nt][0] = 2048 + n * 16 + (((half + key) & 3) << 2);
        bfrIdx[nt][1] = 2048 + n * 16 + (((2 + half + key) & 3) << 2);
    }

    v8f acc[2][4];
#pragma unroll
    for (int i = 0; i < 2; ++i)
#pragma unroll
        for (int j = 0; j < 4; ++j) { v8f z = {}; acc[i][j] = z; }

    // staged tile registers
    uint4 arg[2], blo, bhi;

    auto load_global = [&](int kt) {
        const int k0 = kt * KBLK;
#pragma unroll
        for (int g = 0; g < 2; ++g) {
            if (NSEG == 1) {
                arg[g] = *(const uint4*)(ap0[g] + k0);
            } else {
                const int kk  = k0 + akb;
                const int seg = (kk >= segK) + (kk >= 2 * segK);
                const u16* base = (seg == 0) ? ap0[g] : ((seg == 1) ? ap1[g] : ap2[g]);
                arg[g] = *(const uint4*)(base + (k0 - seg * segK));
            }
        }
        if (FULLK) {
            blo = *(const uint4*)(bp);
            bhi = *(const uint4*)(bp + ldb);
        } else {
            const int kk0 = k0 + 2 * bkp;
            const uint4 z = make_uint4(0u, 0u, 0u, 0u);
            blo = (kk0     < Ktot) ? *(const uint4*)(bp)       : z;
            bhi = (kk0 + 1 < Ktot) ? *(const uint4*)(bp + ldb) : z;
        }
        bp += (size_t)KBLK * ldb;
    };

    auto store_lds = [&](int sel) {
        u32* s = smem + sel;
#pragma unroll
        for (int g = 0; g < 2; ++g) *(uint4*)&s[asIdx[g]] = arg[g];
        const u32 lw[4] = {blo.x, blo.y, blo.z, blo.w};
        const u32 hw[4] = {bhi.x, bhi.y, bhi.z, bhi.w};
#pragma unroll
        for (int j = 0; j < 4; ++j) {
            const u32 a = lw[j], b = hw[j];
            s[bstIdx[j][0]] = (a & 0xFFFFu) | (b << 16);
            s[bstIdx[j][1]] = (a >> 16) | (b & 0xFFFF0000u);
        }
    };

    auto compute = [&](int sel) {
        const u32* s = smem + sel;
        v16bf af[2];
#pragma unroll
        for (int mt = 0; mt < 2; ++mt) {
            v8bf lo = *(const v8bf*)&s[afrIdx[mt][0]];
            v8bf hi = *(const v8bf*)&s[afrIdx[mt][1]];
            af[mt] = __builtin_shufflevector(lo, hi, 0,1,2,3,4,5,6,7,8,9,10,11,12,13,14,15);
        }
        v16bf bfrag[4];
#pragma unroll
        for (int nt = 0; nt < 4; ++nt) {
            v8bf lo = *(const v8bf*)&s[bfrIdx[nt][0]];
            v8bf hi = *(const v8bf*)&s[bfrIdx[nt][1]];
            bfrag[nt] = __builtin_shufflevector(lo, hi, 0,1,2,3,4,5,6,7,8,9,10,11,12,13,14,15);
        }
#pragma unroll
        for (int mt = 0; mt < 2; ++mt)
#pragma unroll
            for (int nt = 0; nt < 4; ++nt)
                acc[mt][nt] = __builtin_amdgcn_wmma_f32_16x16x32_bf16(
                    false, af[mt], false, bfrag[nt], (short)0, acc[mt][nt], false, false);
    };

    const int ntiles = (Ktot + KBLK - 1) / KBLK;

    // ---- software-pipelined, double-buffered main loop ----
    load_global(0);
    store_lds(0);
    __syncthreads();
    int i = 0;
    for (; i + 1 < ntiles; ++i) {
        load_global(i + 1);                    // global latency hidden under WMMAs
        compute((i & 1) * LDSBUF);
        store_lds(((i + 1) & 1) * LDSBUF);
        __syncthreads();                       // one barrier per k-tile
    }
    compute((i & 1) * LDSBUF);

    // ---- epilogue (all branches compile-time) ----
#pragma unroll
    for (int mt = 0; mt < 2; ++mt) {
#pragma unroll
        for (int nt = 0; nt < 4; ++nt) {
            const int gn = colBase + wc * 64 + nt * 16 + lm;
            float bv = 0.0f;
            if (BIAS) bv = bias[gn];
#pragma unroll
            for (int r = 0; r < 8; ++r) {
                const int gm = rowBase + wr * 32 + mt * 16 + 8 * half + r;
                const size_t ci = (size_t)gm * ldc + gn;
                float v = alpha * acc[mt][nt][r] + bv;
                if (CADD == 1) v += cbeta * CaddF[ci];
                if (CADD == 2) v += cbeta * bf2f(CaddB[ci]);
                if (OUTF) Cf[ci] = v;
                if (OUTB) Cb[ci] = f2bf(v);
            }
        }
    }
}

// ---------------- elementwise kernels ----------------

__global__ void zero_kernel(float* __restrict__ p, size_t n) {
    size_t i = (size_t)blockIdx.x * blockDim.x + threadIdx.x;
    if (i < n) p[i] = 0.0f;
}

__global__ void cvt_f32_bf16_kernel(u16* __restrict__ dst, const float* __restrict__ src, size_t n) {
    size_t i = (size_t)blockIdx.x * blockDim.x + threadIdx.x;
    if (i < n) dst[i] = f2bf(src[i]);
}

// W (6, F, O) f32 -> Wb (6, Fp, O) bf16 with zero-padded rows F..Fp-1
__global__ void cvt_w_pad_kernel(u16* __restrict__ dst, const float* __restrict__ src,
                                 int F, int Fp, int O, size_t total)
{
    size_t idx = (size_t)blockIdx.x * blockDim.x + threadIdx.x;
    if (idx >= total) return;
    const int o = (int)(idx % O);
    size_t t = idx / O;
    const int f  = (int)(t % Fp);
    const int sk = (int)(t / Fp);
    const float v = (f < F) ? src[((size_t)sk * F + f) * O + o] : 0.0f;
    dst[idx] = f2bf(v);
}

// Xb(N,B,Fp) = bf16(concat(srcA, state, zero-pad)); gate!=0 -> state*sigmoid(reset)
__global__ void build_x_kernel(u16* __restrict__ Xb,
    const float* __restrict__ srcA, int a_is_bnf, int Fa, int Fp,
    const float* __restrict__ state, const float* __restrict__ gate, size_t total)
{
    size_t idx = (size_t)blockIdx.x * blockDim.x + threadIdx.x;
    if (idx >= total) return;
    const int F = Fa + DC_H;
    const int f = (int)(idx % Fp);
    const size_t nb = idx / Fp;          // n*B + b
    float v = 0.0f;
    if (f < Fa) {
        if (a_is_bnf) {
            const int b = (int)(nb % DC_B);
            const int n = (int)(nb / DC_B);
            v = srcA[((size_t)b * DC_N + n) * Fa + f];
        } else {
            v = srcA[nb * Fa + f];
        }
    } else if (f < F) {
        const int j = f - Fa;
        float s = state[nb * DC_H + j];
        if (gate) {
            const float g = gate[nb * (2 * DC_H) + j];
            s *= 1.0f / (1.0f + expf(-g));
        }
        v = s;
    }
    Xb[idx] = f2bf(v);
}

// state = sigmoid(update)*state + (1-sigmoid(update))*tanh(cand)
__global__ void blend_kernel(float* __restrict__ state,
    const float* __restrict__ gate, const float* __restrict__ cand, size_t total)
{
    size_t idx = (size_t)blockIdx.x * blockDim.x + threadIdx.x;
    if (idx >= total) return;
    const int j = (int)(idx % DC_H);
    const size_t nb = idx / DC_H;
    const float u = 1.0f / (1.0f + expf(-gate[nb * (2 * DC_H) + DC_H + j]));
    const float c = tanhf(cand[idx]);
    state[idx] = u * state[idx] + (1.0f - u) * c;
}

// out(2,B,N,H) <- s0/s1 (N,B,H)
__global__ void out_kernel(float* __restrict__ out,
    const float* __restrict__ s0, const float* __restrict__ s1, size_t total)
{
    size_t idx = (size_t)blockIdx.x * blockDim.x + threadIdx.x;
    if (idx >= total) return;
    const int j = (int)(idx % DC_H);
    size_t t1 = idx / DC_H;
    const int n = (int)(t1 % DC_N);
    size_t t2 = t1 / DC_N;
    const int b = (int)(t2 % DC_B);
    const int l = (int)(t2 / DC_B);
    const float* s = l ? s1 : s0;
    out[idx] = s[((size_t)n * DC_B + b) * DC_H + j];
}

// ---------------- host orchestration ----------------

static void cheb_conv_launch(hipStream_t stream, const u16* Xb, int Fp,
    const u16* Ab, const u16* Wb, const float* bias, float* out, int O,
    u16* T1b, u16* T2b)
{
    const int Ncols = DC_B * Fp;
    const int Kfeat = 3 * Fp;
    const bool fullk = (Kfeat % KBLK) == 0;
    const dim3 gdiff(Ncols / NBLK, DC_N / MBLK, 1);
    const dim3 gfeat(O / NBLK, (DC_N * DC_B) / MBLK, 1);
    for (int s = 0; s < 2; ++s) {
        const u16* A = Ab + (size_t)s * DC_N * DC_N;
        // T1 = bf16(A @ X)
        gemm_ker<1, true, 0, false, true, false><<<gdiff, 256, 0, stream>>>(
            A, A, A, DC_N, DC_N, Xb, Ncols, nullptr, T1b, Ncols,
            1.0f, nullptr, nullptr, 0.0f, nullptr);
        // T2 = bf16(2*A @ T1 - X)
        gemm_ker<1, true, 2, false, true, false><<<gdiff, 256, 0, stream>>>(
            A, A, A, DC_N, DC_N, T1b, Ncols, nullptr, T2b, Ncols,
            2.0f, nullptr, Xb, -1.0f, nullptr);
        // out (+)= [X|T1|T2] @ W[s] ; bias on first support
        const u16* Ws = Wb + (size_t)s * 3 * Fp * O;
        if (s == 0) {
            if (fullk)
                gemm_ker<3, true, 0, true, false, true><<<gfeat, 256, 0, stream>>>(
                    Xb, T1b, T2b, Fp, Kfeat, Ws, O, out, nullptr, O,
                    1.0f, nullptr, nullptr, 0.0f, bias);
            else
                gemm_ker<3, false, 0, true, false, true><<<gfeat, 256, 0, stream>>>(
                    Xb, T1b, T2b, Fp, Kfeat, Ws, O, out, nullptr, O,
                    1.0f, nullptr, nullptr, 0.0f, bias);
        } else {
            if (fullk)
                gemm_ker<3, true, 1, true, false, false><<<gfeat, 256, 0, stream>>>(
                    Xb, T1b, T2b, Fp, Kfeat, Ws, O, out, nullptr, O,
                    1.0f, out, nullptr, 1.0f, nullptr);
            else
                gemm_ker<3, false, 1, true, false, false><<<gfeat, 256, 0, stream>>>(
                    Xb, T1b, T2b, Fp, Kfeat, Ws, O, out, nullptr, O,
                    1.0f, out, nullptr, 1.0f, nullptr);
        }
    }
}

static void run_cell(hipStream_t stream, const float* inp, int a_is_bnf, int Fa, int Fp,
    float* state, const u16* Ab,
    const u16* ruWb, const float* rub, const u16* hWb, const float* hb,
    u16* Xb, u16* T1b, u16* T2b, float* G, float* Cand)
{
    const size_t xTotal = (size_t)DC_N * DC_B * Fp;
    const size_t sTotal = (size_t)DC_N * DC_B * DC_H;
    build_x_kernel<<<(unsigned)((xTotal + 255) / 256), 256, 0, stream>>>(
        Xb, inp, a_is_bnf, Fa, Fp, state, nullptr, xTotal);
    cheb_conv_launch(stream, Xb, Fp, Ab, ruWb, rub, G, 2 * DC_H, T1b, T2b);
    build_x_kernel<<<(unsigned)((xTotal + 255) / 256), 256, 0, stream>>>(
        Xb, inp, a_is_bnf, Fa, Fp, state, G, xTotal);
    cheb_conv_launch(stream, Xb, Fp, Ab, hWb, hb, Cand, DC_H, T1b, T2b);
    blend_kernel<<<(unsigned)((sTotal + 255) / 256), 256, 0, stream>>>(state, G, Cand, sTotal);
}

extern "C" void kernel_launch(void* const* d_in, const int* in_sizes, int n_in,
                              void* d_out, int out_size, void* d_ws, size_t ws_size,
                              hipStream_t stream)
{
    (void)in_sizes; (void)n_in; (void)out_size; (void)ws_size;
    const float* inputs   = (const float*)d_in[0];   // (T,B,N,I)
    const float* supports = (const float*)d_in[1];   // (S,N,N)
    const float* ruW0 = (const float*)d_in[2];
    const float* rub0 = (const float*)d_in[3];
    const float* hW0  = (const float*)d_in[4];
    const float* hb0  = (const float*)d_in[5];
    const float* ruW1 = (const float*)d_in[6];
    const float* rub1 = (const float*)d_in[7];
    const float* hW1  = (const float*)d_in[8];
    const float* hb1  = (const float*)d_in[9];

    const int F0 = DC_I + DC_H;          // 130
    const int F0p = 136;                 // padded: %8==0, 16B-aligned rows
    const int F1 = 2 * DC_H;             // 256
    const size_t NBH  = (size_t)DC_N * DC_B * DC_H;
    const size_t NBF  = (size_t)DC_N * DC_B * (size_t)F1;   // max Fp
    const size_t SNN  = (size_t)2 * DC_N * DC_N;
    const size_t nRu0 = (size_t)6 * F0p * (2 * DC_H);
    const size_t nH0  = (size_t)6 * F0p * DC_H;
    const size_t nRu1 = (size_t)6 * F1 * (2 * DC_H);
    const size_t nH1  = (size_t)6 * F1 * DC_H;
    const size_t SLACK = 256;   // tail slack after K-segmented A buffers

    char* wp = (char*)d_ws;
    auto alloc = [&](size_t bytes) -> char* {
        char* r = wp; wp += (bytes + 255) & ~(size_t)255; return r;
    };
    float* s0   = (float*)alloc(NBH * 4);
    float* s1   = (float*)alloc(NBH * 4);
    float* G    = (float*)alloc(NBF * 4);
    float* Cand = (float*)alloc(NBH * 4);
    u16* Xb  = (u16*)alloc(NBF * 2 + SLACK);
    u16* T1b = (u16*)alloc(NBF * 2 + SLACK);
    u16* T2b = (u16*)alloc(NBF * 2 + SLACK);
    u16* Ab  = (u16*)alloc(SNN * 2);
    u16* Wru0b = (u16*)alloc(nRu0 * 2);
    u16* Wh0b  = (u16*)alloc(nH0 * 2);
    u16* Wru1b = (u16*)alloc(nRu1 * 2);
    u16* Wh1b  = (u16*)alloc(nH1 * 2);

    // per-launch bf16 conversion of loop-invariant operands (W zero-padded to Fp)
    cvt_f32_bf16_kernel<<<(unsigned)((SNN + 255) / 256), 256, 0, stream>>>(Ab, supports, SNN);
    cvt_w_pad_kernel<<<(unsigned)((nRu0 + 255) / 256), 256, 0, stream>>>(Wru0b, ruW0, F0, F0p, 2 * DC_H, nRu0);
    cvt_w_pad_kernel<<<(unsigned)((nH0  + 255) / 256), 256, 0, stream>>>(Wh0b,  hW0,  F0, F0p, DC_H,     nH0);
    cvt_w_pad_kernel<<<(unsigned)((nRu1 + 255) / 256), 256, 0, stream>>>(Wru1b, ruW1, F1, F1,  2 * DC_H, nRu1);
    cvt_w_pad_kernel<<<(unsigned)((nH1  + 255) / 256), 256, 0, stream>>>(Wh1b,  hW1,  F1, F1,  DC_H,     nH1);

    zero_kernel<<<(unsigned)((2 * NBH + 255) / 256), 256, 0, stream>>>(s0, 2 * NBH);

    for (int t = 0; t < DC_T; ++t) {
        const float* x_t = inputs + (size_t)t * DC_B * DC_N * DC_I;
        run_cell(stream, x_t, /*a_is_bnf=*/1, DC_I, F0p, s0, Ab,
                 Wru0b, rub0, Wh0b, hb0, Xb, T1b, T2b, G, Cand);
        run_cell(stream, s0,  /*a_is_bnf=*/0, DC_H, F1, s1, Ab,
                 Wru1b, rub1, Wh1b, hb1, Xb, T1b, T2b, G, Cand);
    }

    const size_t outTotal = (size_t)2 * DC_B * DC_N * DC_H;
    out_kernel<<<(unsigned)((outTotal + 255) / 256), 256, 0, stream>>>(
        (float*)d_out, s0, s1, outTotal);
}
